// GraphAttentionConvLayer_2946347565085
// MI455X (gfx1250) — matrix operand order repaired and checked by reading, not compiled, not run on hardware
//
#include <hip/hip_runtime.h>

#define Bn 8
#define Nn 8192
#define Sn 2048
#define Kn 32
#define RAD2 0.25f
#define EPSc 1e-5f
#define ALPHAc 0.2f

typedef _Float16 h16;
typedef __attribute__((ext_vector_type(16))) _Float16 v16h;
typedef __attribute__((ext_vector_type(8)))  _Float16 v8h;
typedef __attribute__((ext_vector_type(8)))  float    v8f;

__device__ __forceinline__ v16h ld_frag(const h16* p0, const h16* p1) {
  union { v16h v; v8h h[2]; } u;
  u.h[0] = *(const v8h*)p0;   // 16B -> ds_load_b128
  u.h[1] = *(const v8h*)p1;
  return u.v;
}

#define WMMA16(Af, Bf, Cf) \
  __builtin_amdgcn_wmma_f32_16x16x32_f16(false, (Af), false, (Bf), (short)0, (Cf), false, false)

// ---------------------------------------------------------------- BN fold ---
__global__ __launch_bounds__(128)
void bn_fold(const float* cb0, const float* g0, const float* be0, const float* m0, const float* v0,
             const float* cb1, const float* g1, const float* be1, const float* m1, const float* v1,
             const float* cb2, const float* g2, const float* be2, const float* m2, const float* v2,
             float* ws_bn) {
  int c = threadIdx.x;
  if (c < 64) {
    float A = g0[c] / sqrtf(v0[c] + EPSc);
    ws_bn[c]       = A;
    ws_bn[64 + c]  = (cb0[c] - m0[c]) * A + be0[c];
    float A1 = g1[c] / sqrtf(v1[c] + EPSc);
    ws_bn[128 + c] = A1;
    ws_bn[192 + c] = (cb1[c] - m1[c]) * A1 + be1[c];
  }
  if (c < 128) {
    float A2 = g2[c] / sqrtf(v2[c] + EPSc);
    ws_bn[256 + c] = A2;
    ws_bn[384 + c] = (cb2[c] - m2[c]) * A2 + be2[c];
  }
}

// -------------------------------------------------------------------- FPS ---
__global__ __launch_bounds__(1024)
void fps_kernel(const float* xyz, const int* targ, float* out,
                float* ws_nx, int* ws_fps) {
  extern __shared__ char smem[];
  float* lx   = (float*)smem;
  float* ly   = lx + Nn;
  float* lz   = ly + Nn;
  float* dist = lz + Nn;
  int*   cents= (int*)(dist + Nn);
  float* redV = (float*)(cents + Sn);
  int*   redI = (int*)(redV + 32);
  int*   sFar = redI + 32;

  const int b = blockIdx.x, tid = threadIdx.x, nth = blockDim.x;
  const int lane = tid & 31, wid = tid >> 5, nw = nth >> 5;

  for (int t = tid; t < Nn; t += nth) {
    lx[t] = xyz[(b * 3 + 0) * Nn + t];
    ly[t] = xyz[(b * 3 + 1) * Nn + t];
    lz[t] = xyz[(b * 3 + 2) * Nn + t];
    dist[t] = 1e10f;
  }
  if (tid == 0) sFar[0] = 0;
  __syncthreads();

  for (int s = 0; s < Sn; s++) {
    int far = sFar[0];
    if (tid == 0) cents[s] = far;
    float cx = lx[far], cy = ly[far], cz = lz[far];
    float bv = -1.0f; int bi = 0x7fffffff;
    for (int t = tid; t < Nn; t += nth) {
      float dx = lx[t] - cx, dy = ly[t] - cy, dz = lz[t] - cz;
      float d  = dx * dx + dy * dy + dz * dz;
      float nd = fminf(dist[t], d);
      dist[t] = nd;
      if (nd > bv || (nd == bv && t < bi)) { bv = nd; bi = t; }
    }
    for (int off = 16; off; off >>= 1) {
      float v2 = __shfl_xor(bv, off);
      int   i2 = __shfl_xor(bi, off);
      if (v2 > bv || (v2 == bv && i2 < bi)) { bv = v2; bi = i2; }
    }
    if (lane == 0) { redV[wid] = bv; redI[wid] = bi; }
    __syncthreads();
    if (wid == 0) {
      float v = (lane < nw) ? redV[lane] : -1.0f;
      int   i = (lane < nw) ? redI[lane] : 0x7fffffff;
      for (int off = 16; off; off >>= 1) {
        float v2 = __shfl_xor(v, off);
        int   i2 = __shfl_xor(i, off);
        if (v2 > v || (v2 == v && i2 < i)) { v = v2; i = i2; }
      }
      if (lane == 0) sFar[0] = i;
    }
    __syncthreads();
  }

  const size_t tgtBase = (size_t)Bn * 3 * Sn + (size_t)Bn * 128 * Sn;
  for (int s = tid; s < Sn; s += nth) {
    int id = cents[s];
    out[(b * 3 + 0) * Sn + s] = lx[id];
    out[(b * 3 + 1) * Sn + s] = ly[id];
    out[(b * 3 + 2) * Sn + s] = lz[id];
    ws_nx[((size_t)b * Sn + s) * 3 + 0] = lx[id];
    ws_nx[((size_t)b * Sn + s) * 3 + 1] = ly[id];
    ws_nx[((size_t)b * Sn + s) * 3 + 2] = lz[id];
    out[tgtBase + (size_t)b * Sn + s] = (float)targ[b * Nn + id];
    ws_fps[b * Sn + s] = id;
  }
}

// ------------------------------------------------------------- ball query ---
__global__ __launch_bounds__(256)
void ball_kernel(const float* xyz, const float* ws_nx, int* ws_idx) {
  const int tid = threadIdx.x, lane = tid & 31, wv = tid >> 5;
  const int g = blockIdx.x * 8 + wv;          // B*S centers
  const int b = g >> 11;
  float nx0 = ws_nx[g * 3 + 0], nx1 = ws_nx[g * 3 + 1], nx2 = ws_nx[g * 3 + 2];
  int cnt = 0;
  for (int base = 0; base < Nn && cnt < Kn; base += 32) {
    int j = base + lane;
    float dx = xyz[(b * 3 + 0) * Nn + j] - nx0;
    float dy = xyz[(b * 3 + 1) * Nn + j] - nx1;
    float dz = xyz[(b * 3 + 2) * Nn + j] - nx2;
    bool in = (dx * dx + dy * dy + dz * dz) <= RAD2;
    unsigned long long m = __ballot(in);
    int pos = cnt + (int)__popcll(m & ((1ull << lane) - 1ull));
    if (in && pos < Kn) ws_idx[(size_t)g * Kn + pos] = j;
    cnt += (int)__popcll(m);
  }
  if (cnt > Kn) cnt = Kn;
  int first = ws_idx[(size_t)g * Kn];         // center itself always qualifies
  for (int p = cnt + lane; p < Kn; p += 32) ws_idx[(size_t)g * Kn + p] = first;
}

// -------------------------------------------- center MLP + ca = cf @ a_h ----
__global__ __launch_bounds__(256)
void center_kernel(const float* points, const float* w0, const float* w1, const float* w2,
                   const float* amat, const float* ws_bn, const float* ws_nx,
                   const int* ws_fps, float* ws_ca) {
  __shared__ float sA[8][64];
  __shared__ float sB[8][128];
  const int tid = threadIdx.x, lane = tid & 31, wv = tid >> 5;
  const int g = blockIdx.x * 8 + wv;
  const int b = g >> 11;
  const int id = ws_fps[g];

  for (int c = lane; c < 64; c += 32)
    sA[wv][c] = (c < 3) ? ws_nx[(size_t)g * 3 + c]
                        : points[((size_t)b * 61 + (c - 3)) * Nn + id];

  const float* Ws[3] = { w0, w1, w2 };
  const int    coT[3] = { 64, 64, 128 };
  const int    aOff[3] = { 0, 128, 256 };
  float* cur = sA[wv];
  float* nxt = sB[wv];
  for (int l = 0; l < 3; l++) {
    int co = coT[l];
    for (int c = lane; c < co; c += 32) {
      float acc = 0.0f;
      for (int k = 0; k < 64; k++) acc += Ws[l][c * 64 + k] * cur[k];
      float y = acc * ws_bn[aOff[l] + c] + ws_bn[aOff[l] + co + c];
      nxt[c] = fmaxf(y, 0.0f);
    }
    float* t = cur; cur = nxt; nxt = t;
  }
  // cur now holds 128-dim center feature; compute ca = cf @ a[3:,:]
  for (int d = lane; d < 128; d += 32) {
    float acc = 0.0f;
    for (int c = 0; c < 128; c++) acc += cur[c] * amat[(3 + c) * 128 + d];
    ws_ca[(size_t)g * 128 + d] = acc;
  }
}

// ------------------------------------- grouped MLP + attention (WMMA core) --
__global__ __launch_bounds__(128)
void group_kernel(const float* xyz, const float* points,
                  const float* w0, const float* w1, const float* w2,
                  const float* amat, const float* ws_bn, const float* ws_nx,
                  const int* ws_idx, const float* ws_ca, float* out) {
  extern __shared__ char smem[];
  h16*   sW0 = (h16*)smem;            // [64][64]  (co, ci)
  h16*   sW1 = sW0 + 64 * 64;         // [64][64]
  h16*   sW2 = sW1 + 64 * 64;         // [128][64]
  h16*   sAh = sW2 + 128 * 64;        // [128][160]: c<128 -> -a[3+c][d]; 128..130 -> a[c-128][d]; else 0
  float* sBN = (float*)(sAh + 128 * 160);   // 512
  h16*   sTiles = (h16*)(sBN + 512);  // 4 waves * [32][160]

  const int tid = threadIdx.x, lane = tid & 31, wv = tid >> 5;
  const int hi = lane >> 4, lo = lane & 15;

  for (int i = tid; i < 64 * 64; i += 128) { sW0[i] = (h16)w0[i]; sW1[i] = (h16)w1[i]; }
  for (int i = tid; i < 128 * 64; i += 128) sW2[i] = (h16)w2[i];
  for (int i = tid; i < 128 * 160; i += 128) {
    int d = i / 160, c = i - d * 160;
    float v = 0.0f;
    if (c < 128)      v = -amat[(3 + c) * 128 + d];
    else if (c < 131) v =  amat[(c - 128) * 128 + d];
    sAh[i] = (h16)v;
  }
  for (int i = tid; i < 512; i += 128) sBN[i] = ws_bn[i];
  __syncthreads();

  h16* tile = sTiles + (size_t)wv * (32 * 160);
  const h16* Wl[3] = { sW0, sW1, sW2 };
  const size_t outBase = (size_t)Bn * 3 * Sn;

  for (int g = blockIdx.x * 4 + wv; g < Bn * Sn; g += gridDim.x * 4) {
    const int b = g >> 11, s = g & (Sn - 1);

    // ---- build input tile: row = lane (one neighbor), cols 0..63 features,
    //      128..130 delta_p, 131..159 zero
    {
      int id = ws_idx[(size_t)g * Kn + lane];
      float nx0 = ws_nx[(size_t)g * 3 + 0];
      float nx1 = ws_nx[(size_t)g * 3 + 1];
      float nx2 = ws_nx[(size_t)g * 3 + 2];
      float gx0 = xyz[(b * 3 + 0) * Nn + id];
      float gx1 = xyz[(b * 3 + 1) * Nn + id];
      float gx2 = xyz[(b * 3 + 2) * Nn + id];
      h16* row = tile + lane * 160;
      row[0] = (h16)(gx0 - nx0); row[1] = (h16)(gx1 - nx1); row[2] = (h16)(gx2 - nx2);
      for (int c = 0; c < 61; c++)
        row[3 + c] = (h16)points[((size_t)b * 61 + c) * Nn + id];
      row[128] = (h16)(nx0 - gx0); row[129] = (h16)(nx1 - gx1); row[130] = (h16)(nx2 - gx2);
      for (int c = 131; c < 160; c++) row[c] = (h16)0.0f;
    }

    // ---- 3 MLP layers via WMMA (32 x ci -> 32 x co), result back into tile
#pragma unroll
    for (int l = 0; l < 3; l++) {
      const int co = (l == 2) ? 128 : 64;
      const float* Ac = sBN + ((l == 0) ? 0 : (l == 1) ? 128 : 256);
      const float* Bc = Ac + co;
      v16h aF[2][2];
#pragma unroll
      for (int mt = 0; mt < 2; mt++)
#pragma unroll
        for (int kc = 0; kc < 2; kc++) {
          const h16* p = tile + (mt * 16 + lo) * 160 + kc * 32 + hi * 8;
          aF[mt][kc] = ld_frag(p, p + 16);
        }
      const int ntn = co >> 4;
#pragma unroll
      for (int nt = 0; nt < ntn; nt++) {
        int c = nt * 16 + lo;
        v16h bF[2];
#pragma unroll
        for (int kc = 0; kc < 2; kc++) {
          const h16* p = Wl[l] + c * 64 + kc * 32 + hi * 8;
          bF[kc] = ld_frag(p, p + 16);
        }
        float Asc = Ac[c], Bsc = Bc[c];
#pragma unroll
        for (int mt = 0; mt < 2; mt++) {
          v8f acc;
#pragma unroll
          for (int j = 0; j < 8; j++) acc[j] = 0.0f;
          acc = WMMA16(aF[mt][0], bF[0], acc);
          acc = WMMA16(aF[mt][1], bF[1], acc);
#pragma unroll
          for (int j = 0; j < 8; j++) {
            float y = fmaxf(acc[j] * Asc + Bsc, 0.0f);
            tile[(mt * 16 + hi * 8 + j) * 160 + c] = (h16)y;
          }
        }
      }
    }

    // ---- attention: e = ca + dp@a_p - G@a_h  (one 16-col slab per nt)
    for (int nt = 0; nt < 8; nt++) {
      int d = nt * 16 + lo;
      float cad = ws_ca[(size_t)g * 128 + d];
      v8f e0, e1;
#pragma unroll
      for (int j = 0; j < 8; j++) { e0[j] = cad; e1[j] = cad; }
#pragma unroll
      for (int kc = 0; kc < 5; kc++) {
        const h16* pb = sAh + d * 160 + kc * 32 + hi * 8;
        v16h bF = ld_frag(pb, pb + 16);
        const h16* pa0 = tile + lo * 160 + kc * 32 + hi * 8;
        v16h aF0 = ld_frag(pa0, pa0 + 16);
        const h16* pa1 = tile + (16 + lo) * 160 + kc * 32 + hi * 8;
        v16h aF1 = ld_frag(pa1, pa1 + 16);
        e0 = WMMA16(aF0, bF, e0);
        e1 = WMMA16(aF1, bF, e1);
      }
      // leaky relu
#pragma unroll
      for (int j = 0; j < 8; j++) {
        e0[j] = (e0[j] > 0.0f) ? e0[j] : ALPHAc * e0[j];
        e1[j] = (e1[j] > 0.0f) ? e1[j] : ALPHAc * e1[j];
      }
      // softmax over K (32 rows): lane pairs (l, l^16) cover all rows of column d
      float mx = -3.4e38f;
#pragma unroll
      for (int j = 0; j < 8; j++) mx = fmaxf(mx, fmaxf(e0[j], e1[j]));
      mx = fmaxf(mx, __shfl_xor(mx, 16));
      float p0[8], p1[8], sum = 0.0f;
#pragma unroll
      for (int j = 0; j < 8; j++) {
        p0[j] = __expf(e0[j] - mx);
        p1[j] = __expf(e1[j] - mx);
        sum += p0[j] + p1[j];
      }
      sum += __shfl_xor(sum, 16);
      float inv = 1.0f / sum;
      float o = 0.0f;
#pragma unroll
      for (int j = 0; j < 8; j++) {
        int r0 = hi * 8 + j, r1 = 16 + hi * 8 + j;
        o += p0[j] * (float)tile[r0 * 160 + d];
        o += p1[j] * (float)tile[r1 * 160 + d];
      }
      o *= inv;
      o += __shfl_xor(o, 16);
      if (lane < 16)
        out[outBase + ((size_t)b * 128 + d) * Sn + s] = o;
    }
  }
}

// ------------------------------------------------------------------- host ---
extern "C" void kernel_launch(void* const* d_in, const int* in_sizes, int n_in,
                              void* d_out, int out_size, void* d_ws, size_t ws_size,
                              hipStream_t stream) {
  const float* xyz    = (const float*)d_in[0];
  const float* points = (const float*)d_in[1];
  const int*   targ   = (const int*)d_in[2];
  const float* w0  = (const float*)d_in[3];
  const float* cb0 = (const float*)d_in[4];
  const float* g0  = (const float*)d_in[5];
  const float* be0 = (const float*)d_in[6];
  const float* m0  = (const float*)d_in[7];
  const float* v0  = (const float*)d_in[8];
  const float* w1  = (const float*)d_in[9];
  const float* cb1 = (const float*)d_in[10];
  const float* g1  = (const float*)d_in[11];
  const float* be1 = (const float*)d_in[12];
  const float* m1  = (const float*)d_in[13];
  const float* v1  = (const float*)d_in[14];
  const float* w2  = (const float*)d_in[15];
  const float* cb2 = (const float*)d_in[16];
  const float* g2  = (const float*)d_in[17];
  const float* be2 = (const float*)d_in[18];
  const float* m2  = (const float*)d_in[19];
  const float* v2  = (const float*)d_in[20];
  const float* amat= (const float*)d_in[21];
  float* out = (float*)d_out;

  float* wsf    = (float*)d_ws;
  float* ws_nx  = wsf;                                        // B*S*3
  int*   ws_fps = (int*)(wsf + (size_t)Bn * Sn * 3);          // B*S
  int*   ws_idx = ws_fps + (size_t)Bn * Sn;                   // B*S*K
  float* ws_ca  = (float*)(ws_idx + (size_t)Bn * Sn * Kn);    // B*S*128
  float* ws_bn  = ws_ca + (size_t)Bn * Sn * 128;              // 512

  bn_fold<<<1, 128, 0, stream>>>(cb0, g0, be0, m0, v0,
                                 cb1, g1, be1, m1, v1,
                                 cb2, g2, be2, m2, v2, ws_bn);

  size_t fpsLds = (size_t)4 * Nn * 4 + (size_t)Sn * 4 + 64 * 4 + 16;
  fps_kernel<<<Bn, 1024, fpsLds, stream>>>(xyz, targ, out, ws_nx, ws_fps);

  ball_kernel<<<(Bn * Sn) / 8, 256, 0, stream>>>(xyz, ws_nx, ws_idx);

  center_kernel<<<(Bn * Sn) / 8, 256, 0, stream>>>(points, w0, w1, w2, amat,
                                                   ws_bn, ws_nx, ws_fps, ws_ca);

  size_t grpLds = (size_t)(64 * 64 + 64 * 64 + 128 * 64 + 128 * 160) * 2
                + 512 * 4 + (size_t)4 * 32 * 160 * 2;
  group_kernel<<<1024, 128, grpLds, stream>>>(xyz, points, w0, w1, w2, amat,
                                              ws_bn, ws_nx, ws_idx, ws_ca, out);
}
